// MultiHead_16509854286019
// MI455X (gfx1250) — matrix operand order repaired and checked by reading, not compiled, not run on hardware
//
#include <hip/hip_runtime.h>
#include <hip/hip_bf16.h>

typedef __attribute__((ext_vector_type(16))) _Float16 v16h;
typedef __attribute__((ext_vector_type(8)))  _Float16 v8h;
typedef __attribute__((ext_vector_type(8)))  float    v8f;

#define BATCH 4
#define T_SEQ 2048
#define EMB   1024
#define NH    16
#define HS    64

// D = A(16x32 f16) * B(32x16 f16) + C(16x16 f32)
__device__ __forceinline__ v8f wmma16(v16h a, v16h b, v8f c) {
  return __builtin_amdgcn_wmma_f32_16x16x32_f16(false, a, false, b, (short)0, c,
                                                false, false);
}

// ---------------------------------------------------------------------------
// Kernel A: repack Wq/Wk/Wv (fp32 [H][E][64]) into f16 B-fragment order.
// Fragment index f = (h*32 + kstep)*12 + mat*4 + chunk; storage [f][lane][j].
// B-fragment order used here (32x16 f16, contiguous k, same order as the A loads): lane l -> col n=l&15, half hb=l>>4;
//   element j (0..15) = B[hb*16 + j][n]  => W[e = kstep*32 + hb*16 + j][d = chunk*16 + n]
// ---------------------------------------------------------------------------
__global__ void swizzle_w_kernel(const float* __restrict__ Wq,
                                 const float* __restrict__ Wk,
                                 const float* __restrict__ Wv,
                                 _Float16* __restrict__ W16) {
  int idx = blockIdx.x * blockDim.x + threadIdx.x;   // < 3,145,728
  int j = idx & 15;
  int t = idx >> 4;
  int lane = t & 31;  t >>= 5;
  int chunk = t & 3;  t >>= 2;
  int mat = t % 3;    t /= 3;
  int kstep = t & 31;
  int h = t >> 5;
  if (h >= NH) return;
  const float* W = (mat == 0) ? Wq : (mat == 1) ? Wk : Wv;
  int e = kstep * 32 + (lane >> 4) * 16 + j;
  int d = chunk * 16 + (lane & 15);
  W16[idx] = (_Float16)W[(size_t)h * EMB * HS + (size_t)e * HS + d];
}

// ---------------------------------------------------------------------------
// Kernel B: QKV projection. One wave handles one 16-row tile for one head.
// acc[mat*4 + chunk] : 12 x (16x16 f32) output tiles, K-dim = 1024 in 32 steps.
// All 12 B-fragments are loaded as one batch (clause of b128 loads) before the
// WMMA chain so the matrix pipe overlaps with in-flight loads.
// Q,K stored f16 row-major [b][h][t][64]; V stored transposed [b][h][64][t].
// ---------------------------------------------------------------------------
__global__ __launch_bounds__(32) void qkv_proj_kernel(
    const float*    __restrict__ x,
    const _Float16* __restrict__ W16,
    _Float16* __restrict__ Q16,
    _Float16* __restrict__ K16,
    _Float16* __restrict__ Vt16) {
  const int tile = blockIdx.x;     // 0..511 over B*T/16 (never straddles batch)
  const int h    = blockIdx.y;     // 0..15
  const int lane = threadIdx.x;
  const int n    = lane & 15;
  const int hb   = lane >> 4;
  const int r0   = tile * 16;      // row in flattened [B*T]

  v8f acc[12];
#pragma unroll
  for (int i = 0; i < 12; ++i) acc[i] = (v8f)0.0f;

  const float* xrow = x + (size_t)(r0 + n) * EMB;  // A-matrix row m = lane&15
  for (int kstep = 0; kstep < 32; ++kstep) {
    // A-fragment: two contiguous 8-float runs per lane, cvt fp32 -> f16
    v8f x0 = *(const v8f*)(xrow + kstep * 32 + hb * 16);
    v8f x1 = *(const v8f*)(xrow + kstep * 32 + hb * 16 + 8);
    const _Float16* wp =
        W16 + ((size_t)(h * 32 + kstep) * 12 * 32 + lane) * 16;
    // batch all 12 B-fragment loads (independent) before any WMMA
    v16h bf[12];
#pragma unroll
    for (int mc = 0; mc < 12; ++mc)
      bf[mc] = *(const v16h*)(wp + (size_t)mc * 512);
    v16h a;
#pragma unroll
    for (int i = 0; i < 8; ++i) {
      a[i]     = (_Float16)x0[i];
      a[8 + i] = (_Float16)x1[i];
    }
#pragma unroll
    for (int mc = 0; mc < 12; ++mc)
      acc[mc] = wmma16(a, bf[mc], acc[mc]);
  }

  const int b   = r0 / T_SEQ;
  const int t0l = r0 % T_SEQ;
  const int bh  = b * NH + h;
  const size_t qk_base = ((size_t)bh * T_SEQ + t0l) * HS;
  // C/D layout: lane (n, hb), VGPR r -> element (row 8*hb + r, col n)
#pragma unroll
  for (int c = 0; c < 4; ++c) {
#pragma unroll
    for (int r = 0; r < 8; ++r) {
      Q16[qk_base + (size_t)(hb * 8 + r) * HS + c * 16 + n] = (_Float16)acc[c][r];
      K16[qk_base + (size_t)(hb * 8 + r) * HS + c * 16 + n] = (_Float16)acc[4 + c][r];
    }
    v8h vv;
#pragma unroll
    for (int r = 0; r < 8; ++r) vv[r] = (_Float16)acc[8 + c][r];
    *(v8h*)(Vt16 + ((size_t)bh * HS + c * 16 + n) * T_SEQ + t0l + hb * 8) = vv;
  }
}

// ---------------------------------------------------------------------------
// Kernel C: causal flash attention, one wave per (b, h, 16-query tile).
// Scores via WMMA(Q, K^T), online softmax in fp32, P bounced through LDS to
// A-layout, output via WMMA(P, V) against transposed V (contiguous B-frags).
// K / V fragments are batch-loaded before each WMMA group.
// scale = 1/sqrt(E) = 1/32 (reference scales by sqrt(embed), not head size).
// ---------------------------------------------------------------------------
__global__ __launch_bounds__(32) void attn_kernel(
    const _Float16* __restrict__ Q16,
    const _Float16* __restrict__ K16,
    const _Float16* __restrict__ Vt16,
    float* __restrict__ out) {
  __shared__ _Float16 ldsP[16 * 32];
  const int qt   = blockIdx.x;   // 0..127
  const int bh   = blockIdx.y;   // 0..63
  const int b    = bh >> 4;
  const int h    = bh & 15;
  const int lane = threadIdx.x;
  const int n    = lane & 15;
  const int hb   = lane >> 4;
  const int t0   = qt * 16;

  // Q A-fragments: d-chunks kb=0,1 (K=32 each)
  v16h qa[2];
  const _Float16* qrow = Q16 + ((size_t)bh * T_SEQ + t0 + n) * HS;
#pragma unroll
  for (int kb = 0; kb < 2; ++kb) {
    v8h q0 = *(const v8h*)(qrow + kb * 32 + hb * 16);
    v8h q1 = *(const v8h*)(qrow + kb * 32 + hb * 16 + 8);
#pragma unroll
    for (int i = 0; i < 8; ++i) { qa[kb][i] = q0[i]; qa[kb][8 + i] = q1[i]; }
  }

  float mrun[8], lrun[8];
  v8f oacc[4];
#pragma unroll
  for (int r = 0; r < 8; ++r) { mrun[r] = -3.0e38f; lrun[r] = 0.0f; }
#pragma unroll
  for (int c = 0; c < 4; ++c) oacc[c] = (v8f)0.0f;

  const float scale = 0.03125f;             // 1/sqrt(1024)
  const int nblk = (t0 + 47) >> 5;          // ceil((t0+16)/32) causal key blocks

  for (int kblk = 0; kblk < nblk; ++kblk) {
    const int ks0 = kblk * 32;
    // ---- batch-load 4 K B-fragments, then 4 score WMMAs (S = two 16x16 tiles)
    const _Float16* kbase = K16 + ((size_t)bh * T_SEQ + ks0 + n) * HS + hb * 16;
    v16h kf[2][2];
#pragma unroll
    for (int kb = 0; kb < 2; ++kb) {
      kf[0][kb] = *(const v16h*)(kbase + kb * 32);
      kf[1][kb] = *(const v16h*)(kbase + (size_t)16 * HS + kb * 32);
    }
    v8f s0 = (v8f)0.0f, s1 = (v8f)0.0f;
#pragma unroll
    for (int kb = 0; kb < 2; ++kb) {
      s0 = wmma16(qa[kb], kf[0][kb], s0);
      s1 = wmma16(qa[kb], kf[1][kb], s1);
    }
    // ---- prefetch V B-fragments for this block while softmax runs
    v16h vf[4];
#pragma unroll
    for (int c = 0; c < 4; ++c)
      vf[c] = *(const v16h*)(Vt16 + ((size_t)bh * HS + c * 16 + n) * T_SEQ +
                             ks0 + hb * 16);
    // ---- online softmax (per query row; rows split across VGPR r + lane half)
#pragma unroll
    for (int r = 0; r < 8; ++r) {
      const int trow = t0 + hb * 8 + r;
      float v0 = s0[r] * scale;
      float v1 = s1[r] * scale;
      if (ks0 + n > trow)      v0 = -3.0e38f;   // causal mask
      if (ks0 + 16 + n > trow) v1 = -3.0e38f;
      float mx = fmaxf(v0, v1);
#pragma unroll
      for (int off = 1; off < 16; off <<= 1)    // reduce within 16-lane half
        mx = fmaxf(mx, __shfl_xor(mx, off, 32));
      float mnew  = fmaxf(mrun[r], mx);
      float alpha = __expf(mrun[r] - mnew);
      mrun[r] = mnew;
      float p0 = __expf(v0 - mnew);
      float p1 = __expf(v1 - mnew);
      float sm = p0 + p1;
#pragma unroll
      for (int off = 1; off < 16; off <<= 1)
        sm += __shfl_xor(sm, off, 32);
      lrun[r] = lrun[r] * alpha + sm;
      ldsP[(hb * 8 + r) * 32 + n]      = (_Float16)p0;
      ldsP[(hb * 8 + r) * 32 + 16 + n] = (_Float16)p1;
#pragma unroll
      for (int c = 0; c < 4; ++c) oacc[c][r] *= alpha;
    }
    __syncthreads();
    // ---- P: C/D layout -> A layout via LDS
    v16h pa;
    {
      const _Float16* pr = ldsP + n * 32;     // A row m = lane&15
      v8h p0v = *(const v8h*)(pr + hb * 16);
      v8h p1v = *(const v8h*)(pr + hb * 16 + 8);
#pragma unroll
      for (int i = 0; i < 8; ++i) { pa[i] = p0v[i]; pa[8 + i] = p1v[i]; }
    }
    // ---- O += P * V
#pragma unroll
    for (int c = 0; c < 4; ++c)
      oacc[c] = wmma16(pa, vf[c], oacc[c]);
    __syncthreads();   // protect LDS before next block overwrites it
  }

  // ---- normalize + store, heads concatenated: out[b][t][h*64 + d]
#pragma unroll
  for (int c = 0; c < 4; ++c) {
#pragma unroll
    for (int r = 0; r < 8; ++r) {
      const int trow = t0 + hb * 8 + r;
      out[((size_t)b * T_SEQ + trow) * EMB + h * HS + c * 16 + n] =
          oacc[c][r] / lrun[r];
    }
  }
}

extern "C" void kernel_launch(void* const* d_in, const int* in_sizes, int n_in,
                              void* d_out, int out_size, void* d_ws, size_t ws_size,
                              hipStream_t stream) {
  (void)in_sizes; (void)n_in; (void)out_size; (void)ws_size;
  const float* x  = (const float*)d_in[0];
  const float* Wq = (const float*)d_in[1];
  const float* Wk = (const float*)d_in[2];
  const float* Wv = (const float*)d_in[3];
  float* out = (float*)d_out;

  // Workspace layout (f16):
  //   W16  : 3*H*E*HS           = 3,145,728 halves (6 MB)
  //   Q16  : B*H*T*HS           = 8,388,608 halves (16 MB)
  //   K16  : same               (16 MB)
  //   Vt16 : same, transposed   (16 MB)
  _Float16* W16  = (_Float16*)d_ws;
  _Float16* Q16  = W16 + (size_t)3 * NH * EMB * HS;
  _Float16* K16  = Q16 + (size_t)BATCH * NH * T_SEQ * HS;
  _Float16* Vt16 = K16 + (size_t)BATCH * NH * T_SEQ * HS;

  const int swz_elems = 3 * NH * EMB * HS;  // 3,145,728
  swizzle_w_kernel<<<dim3(swz_elems / 256), 256, 0, stream>>>(Wq, Wk, Wv, W16);
  qkv_proj_kernel<<<dim3((BATCH * T_SEQ) / 16, NH), 32, 0, stream>>>(
      x, W16, Q16, K16, Vt16);
  attn_kernel<<<dim3(T_SEQ / 16, BATCH * NH), 32, 0, stream>>>(
      Q16, K16, Vt16, out);
}
